// GbGraphConvNet_17918603558965
// MI455X (gfx1250) — matrix operand order repaired and checked by reading, not compiled, not run on hardware
//
#include <hip/hip_runtime.h>
#include <hip/hip_bf16.h>
#include <math.h>

typedef __attribute__((ext_vector_type(16))) _Float16 v16h;
typedef __attribute__((ext_vector_type(8)))  float    v8f;

#define N_NODES  50000
#define N_EDGES  1000000
#define N_GRAPHS 256
#define ORIG     92
#define ATOMF    64
#define NBR      41
#define KTOT     169   // 64 + 64 + 41
#define NCOL     128
#define EPSBN    1e-5f
#define EDGES_PER_BLOCK 128   // 8 waves * 16 edges

// ---------------- workspace layout (in floats) ----------------
#define WFC_HALVES  24576
#define ATOM_OFF    12288
#define SUM_OFF     (ATOM_OFF + N_NODES*ATOMF)
#define H_OFF       (SUM_OFF  + N_NODES*ATOMF)
#define LG_OFF      (H_OFF    + N_NODES*ATOMF)
#define EV_OFF      (LG_OFF   + N_NODES)
#define ACC_OFF     (EV_OFF   + N_NODES)
#define A_GSUM      (ACC_OFF)
#define A_GSQ       (A_GSUM + 128)
#define A_SC1       (A_GSQ  + 128)
#define A_SH1       (A_SC1  + 128)
#define A_SUM2      (A_SH1  + 128)
#define A_SQ2       (A_SUM2 + 64)
#define A_SC2       (A_SQ2  + 64)
#define A_SH2       (A_SC2  + 64)
#define A_SEGMAX    (A_SH2  + 64)
#define A_DENOM     (A_SEGMAX + 256)
#define A_CRYS      (A_DENOM  + 256)
#define ACC_COUNT   (128*4 + 64*4 + 256 + 256 + 256*64)

__device__ __forceinline__ unsigned encf(float f) {
    unsigned u = __float_as_uint(f);
    return (u & 0x80000000u) ? ~u : (u | 0x80000000u);
}
__device__ __forceinline__ float decf(unsigned u) {
    return (u & 0x80000000u) ? __uint_as_float(u & 0x7fffffffu) : __uint_as_float(~u);
}

#define CVT8(dst, base, q0, q1)                                              \
    dst[base+0] = (_Float16)q0.x; dst[base+1] = (_Float16)q0.y;              \
    dst[base+2] = (_Float16)q0.z; dst[base+3] = (_Float16)q0.w;              \
    dst[base+4] = (_Float16)q1.x; dst[base+5] = (_Float16)q1.y;              \
    dst[base+6] = (_Float16)q1.z; dst[base+7] = (_Float16)q1.w;

// ---------------- K0: zero accumulators ----------------
__global__ void k0_init(float* ws) {
    long i = (long)blockIdx.x * blockDim.x + threadIdx.x;
    long nsum = (long)N_NODES * ATOMF;
    if (i < nsum) ws[SUM_OFF + i] = 0.f;
    else {
        long j = i - nsum;
        if (j < ACC_COUNT) ws[ACC_OFF + j] = 0.f;   // segmax-as-uint 0 == sentinel
    }
}

// ---------------- K1: atom_fea = x@emb_W + emb_b + lower_f ----------------
__global__ void k1_node_embed(const float* __restrict__ x,
                              const float* __restrict__ lower_f,
                              const float* __restrict__ emb_W,
                              const float* __restrict__ emb_b,
                              float* __restrict__ ws) {
    __shared__ float xs[ORIG];
    int node = blockIdx.x, t = threadIdx.x;
    if (t < ORIG) xs[t] = x[(long)node * ORIG + t];
    __syncthreads();
    if (t < ATOMF) {
        float acc = emb_b[t] + lower_f[(long)node * ATOMF + t];
        #pragma unroll 4
        for (int k = 0; k < ORIG; ++k) acc = fmaf(xs[k], emb_W[k * ATOMF + t], acc);
        ws[ATOM_OFF + (long)node * ATOMF + t] = acc;
    }
}

// ---------------- K1b: pre-swizzle fc_W into f16 B-fragment order ----------------
// flat idx = ((c*8+nt)*32 + lane)*16 + j ; element (k = 32c + 8*(lane>=16) + (j&7) + 16*(j>>3), n = 16nt + lane&15)
__global__ void k1b_convW(const float* __restrict__ fc_W, float* ws) {
    int idx = blockIdx.x * blockDim.x + threadIdx.x;
    if (idx >= WFC_HALVES) return;
    int j    = idx & 15;
    int lane = (idx >> 4) & 31;
    int nt   = (idx >> 9) & 7;
    int c    = idx >> 12;
    int k = 32 * c + 8 * (lane >> 4) + (j & 7) + 16 * (j >> 3);
    int n = 16 * nt + (lane & 15);
    float v = (k < KTOT) ? fc_W[k * NCOL + n] : 0.f;
    ((_Float16*)ws)[idx] = (_Float16)v;
}

// ---------------- shared GEMM core: one wave computes 16 edges x 128 cols ----------------
// A fragment mapping (16-bit A 16x32): lane holds row m=lane&15;
// halves j=0..7 -> k = 32c + 8*hi + j ; j=8..15 -> k = 32c + 8*hi + 16 + (j-8)
// Every 8-run is contiguous in k and lies in one source region.
__device__ __forceinline__ void edge_gemm_tile(
    const float* __restrict__ atom_fea, const float* __restrict__ edge_attr,
    const int* __restrict__ src, const int* __restrict__ dst,
    const _Float16* ldsW, long tb, int lane, v8f acc[8]) {
    int m  = lane & 15;
    int hi = lane >> 4;
    int h8 = hi * 8;
    long e = tb + m;
    const float* rowS = atom_fea + (long)src[e] * ATOMF;
    const float* rowD = atom_fea + (long)dst[e] * ATOMF;
    const float* rowE = edge_attr + e * NBR;

    // chunks 0..3: fully inside src (c<2) or dst (c>=2) feature rows; 16B vector loads
    #pragma unroll
    for (int c = 0; c < 4; ++c) {
        const float* row = (c < 2) ? rowS : rowD;
        int off = (c & 1) * 32 + h8;            // run0 start within 64-float row
        float4 q0 = *(const float4*)(row + off);
        float4 q1 = *(const float4*)(row + off + 4);
        float4 q2 = *(const float4*)(row + off + 16);
        float4 q3 = *(const float4*)(row + off + 20);
        v16h a;
        CVT8(a, 0, q0, q1);
        CVT8(a, 8, q2, q3);
        #pragma unroll
        for (int nt = 0; nt < 8; ++nt) {
            v16h b = *(const v16h*)&ldsW[((c * 8 + nt) * 32 + lane) * 16];
            acc[nt] = __builtin_amdgcn_wmma_f32_16x16x32_f16(
                false, a, false, b, (short)0, acc[nt], false, false);
        }
    }
    // chunk 4: edge_attr cols [h8 .. h8+7] and [16+h8 .. 23+h8], all < 32 < NBR.
    // edge_attr rows have stride 41 floats (only 4B alignment) -> dword loads.
    {
        float fe[16];
        #pragma unroll
        for (int i = 0; i < 8; ++i) fe[i]     = rowE[h8 + i];
        #pragma unroll
        for (int i = 0; i < 8; ++i) fe[8 + i] = rowE[16 + h8 + i];
        v16h a;
        #pragma unroll
        for (int i = 0; i < 16; ++i) a[i] = (_Float16)fe[i];
        #pragma unroll
        for (int nt = 0; nt < 8; ++nt) {
            v16h b = *(const v16h*)&ldsW[((4 * 8 + nt) * 32 + lane) * 16];
            acc[nt] = __builtin_amdgcn_wmma_f32_16x16x32_f16(
                false, a, false, b, (short)0, acc[nt], false, false);
        }
    }
    // chunk 5: run0 = edge_attr cols [32+h8 .. 39+h8] (tail mask via clamped addr + select),
    //          run1 = all zero padding.
    {
        v16h a;
        #pragma unroll
        for (int i = 0; i < 8; ++i) {
            int o = 32 + h8 + i;
            float v = rowE[(o < NBR) ? o : (NBR - 1)];   // clamped, always in-bounds
            a[i] = (_Float16)((o < NBR) ? v : 0.f);
        }
        #pragma unroll
        for (int i = 8; i < 16; ++i) a[i] = (_Float16)0.f;
        #pragma unroll
        for (int nt = 0; nt < 8; ++nt) {
            v16h b = *(const v16h*)&ldsW[((5 * 8 + nt) * 32 + lane) * 16];
            acc[nt] = __builtin_amdgcn_wmma_f32_16x16x32_f16(
                false, a, false, b, (short)0, acc[nt], false, false);
        }
    }
}

// ---------------- K2: GEMM pass 1 -> column sum / sumsq of z ----------------
__global__ void k2_gemm_stats(const float* __restrict__ edge_attr,
                              const int* __restrict__ ei,
                              float* __restrict__ ws) {
    __shared__ _Float16 ldsW[WFC_HALVES];
    __shared__ float lsum[NCOL], lsq[NCOL];
    const float* atom_fea = ws + ATOM_OFF;
    const int* src = ei;
    const int* dst = ei + N_EDGES;
    int t = threadIdx.x;
    {   // stage B into LDS
        const uint4* s4 = (const uint4*)ws;
        uint4* d4 = (uint4*)ldsW;
        for (int i = t; i < WFC_HALVES / 8; i += 256) d4[i] = s4[i];
        if (t < NCOL) { lsum[t] = 0.f; lsq[t] = 0.f; }
    }
    __syncthreads();

    int wave = t >> 5, lane = t & 31;
    long tb = (long)blockIdx.x * EDGES_PER_BLOCK + wave * 16;
    if (tb < N_EDGES) {                      // wave-uniform: tile fully valid
        v8f acc[8];
        #pragma unroll
        for (int i = 0; i < 8; ++i) acc[i] = (v8f){0.f,0.f,0.f,0.f,0.f,0.f,0.f,0.f};
        edge_gemm_tile(atom_fea, edge_attr, src, dst, ldsW, tb, lane, acc);
        #pragma unroll
        for (int nt = 0; nt < 8; ++nt) {
            int n = nt * 16 + (lane & 15);
            float p = 0.f, q = 0.f;
            #pragma unroll
            for (int r = 0; r < 8; ++r) { float v = acc[nt][r]; p += v; q += v * v; }
            atomicAdd(&lsum[n], p);
            atomicAdd(&lsq[n], q);
        }
    }
    __syncthreads();
    if (t < NCOL) {
        atomicAdd(&ws[A_GSUM + t], lsum[t]);
        atomicAdd(&ws[A_GSQ  + t], lsq[t]);
    }
}

// ---------------- K3: finalize BN1 (fc_b cancels inside BN) ----------------
__global__ void k3_fin1(const float* __restrict__ g1, const float* __restrict__ b1,
                        float* ws) {
    int n = threadIdx.x;
    if (n >= NCOL) return;
    float mu  = ws[A_GSUM + n] * (1.f / (float)N_EDGES);
    float var = ws[A_GSQ  + n] * (1.f / (float)N_EDGES) - mu * mu;
    float sc  = g1[n] * rsqrtf(var + EPSBN);
    ws[A_SC1 + n] = sc;
    ws[A_SH1 + n] = b1[n] - mu * sc;
}

// ---------------- K4: GEMM pass 2 -> BN -> sigmoid*softplus -> scatter-add ----------------
__global__ void k4_gemm_msg(const float* __restrict__ edge_attr,
                            const int* __restrict__ ei,
                            float* __restrict__ ws) {
    __shared__ _Float16 ldsW[WFC_HALVES];
    const float* atom_fea = ws + ATOM_OFF;
    const int* src = ei;
    const int* dst = ei + N_EDGES;
    int t = threadIdx.x;
    {
        const uint4* s4 = (const uint4*)ws;
        uint4* d4 = (uint4*)ldsW;
        for (int i = t; i < WFC_HALVES / 8; i += 256) d4[i] = s4[i];
    }
    __syncthreads();

    int wave = t >> 5, lane = t & 31, hi = lane >> 4;
    long tb = (long)blockIdx.x * EDGES_PER_BLOCK + wave * 16;
    if (tb >= N_EDGES) return;               // wave-uniform

    v8f acc[8];
    #pragma unroll
    for (int i = 0; i < 8; ++i) acc[i] = (v8f){0.f,0.f,0.f,0.f,0.f,0.f,0.f,0.f};
    edge_gemm_tile(atom_fea, edge_attr, src, dst, ldsW, tb, lane, acc);

    float* summed = ws + SUM_OFF;
    int ln = lane & 15;
    int se[8];
    #pragma unroll
    for (int r = 0; r < 8; ++r) se[r] = src[tb + r + 8 * hi];
    #pragma unroll
    for (int nt = 0; nt < 4; ++nt) {
        int nf = nt * 16 + ln;
        int nc = nf + 64;
        float sf = ws[A_SC1 + nf], hf = ws[A_SH1 + nf];
        float sc = ws[A_SC1 + nc], hc = ws[A_SH1 + nc];
        #pragma unroll
        for (int r = 0; r < 8; ++r) {
            float f  = acc[nt    ][r] * sf + hf;
            float cr = acc[nt + 4][r] * sc + hc;
            float sg = 1.f / (1.f + __expf(-f));
            float sp = fmaxf(cr, 0.f) + log1pf(__expf(-fabsf(cr)));
            atomicAdd(&summed[(long)se[r] * ATOMF + nf], sg * sp);
        }
    }
}

// ---------------- K5: column stats of summed ----------------
__global__ void k5_stats2(float* ws) {
    int t = threadIdx.x;                 // column 0..63
    long r0 = (long)blockIdx.x * 256;
    float p = 0.f, q = 0.f;
    for (long r = r0; r < r0 + 256 && r < N_NODES; ++r) {
        float v = ws[SUM_OFF + r * ATOMF + t];
        p += v; q += v * v;
    }
    atomicAdd(&ws[A_SUM2 + t], p);
    atomicAdd(&ws[A_SQ2  + t], q);
}

// ---------------- K6: finalize BN2 ----------------
__global__ void k6_fin2(const float* __restrict__ g2, const float* __restrict__ b2,
                        float* ws) {
    int n = threadIdx.x;
    if (n >= ATOMF) return;
    float mu  = ws[A_SUM2 + n] * (1.f / (float)N_NODES);
    float var = ws[A_SQ2  + n] * (1.f / (float)N_NODES) - mu * mu;
    float sc  = g2[n] * rsqrtf(var + EPSBN);
    ws[A_SC2 + n] = sc;
    ws[A_SH2 + n] = b2[n] - mu * sc;
}

// ---------------- K7: BN2 -> atom_out(d_out) -> silu -> logits -> segment max ----------------
__global__ void k7_node(const float* __restrict__ att_W, const float* __restrict__ att_b,
                        const int* __restrict__ batch,
                        float* __restrict__ ws, float* __restrict__ out) {
    __shared__ float red[ATOMF];
    int node = blockIdx.x, t = threadIdx.x;
    float s = ws[SUM_OFF + (long)node * ATOMF + t];
    float a = s * ws[A_SC2 + t] + ws[A_SH2 + t];
    out[(long)node * ATOMF + t] = a;                 // atom_out
    float hv = a / (1.f + __expf(-a));               // silu
    ws[H_OFF + (long)node * ATOMF + t] = hv;
    red[t] = hv * att_W[t];
    __syncthreads();
    #pragma unroll
    for (int sft = 32; sft > 0; sft >>= 1) {
        if (t < sft) red[t] += red[t + sft];
        __syncthreads();
    }
    if (t == 0) {
        float lg = red[0] + att_b[0];
        ws[LG_OFF + node] = lg;
        atomicMax((unsigned*)&ws[A_SEGMAX + batch[node]], encf(lg));
    }
}

// ---------------- K8: e = exp(logit - segmax), denom accumulation ----------------
__global__ void k8_expsum(const int* __restrict__ batch, float* ws) {
    int i = blockIdx.x * blockDim.x + threadIdx.x;
    if (i >= N_NODES) return;
    int b = batch[i];
    float m = decf(((const unsigned*)ws)[A_SEGMAX + b]);
    float e = __expf(ws[LG_OFF + i] - m);
    ws[EV_OFF + i] = e;
    atomicAdd(&ws[A_DENOM + b], e);
}

// ---------------- K9: crys += alpha * h ----------------
__global__ void k9_pool(const int* __restrict__ batch, float* ws) {
    long idx = (long)blockIdx.x * blockDim.x + threadIdx.x;
    if (idx >= (long)N_NODES * ATOMF) return;
    int i = (int)(idx >> 6);
    int n = (int)(idx & 63);
    int b = batch[i];
    float alpha = ws[EV_OFF + i] / ws[A_DENOM + b];
    atomicAdd(&ws[A_CRYS + b * ATOMF + n], alpha * ws[H_OFF + idx]);
}

// ---------------- K10: crys_fea = crys @ out_W + out_b ----------------
__global__ void k10_out(const float* __restrict__ out_W, const float* __restrict__ out_b,
                        const float* __restrict__ ws, float* __restrict__ out) {
    int g = threadIdx.x;   // one block, 256 threads
    float acc = 0.f;
    #pragma unroll 4
    for (int n = 0; n < ATOMF; ++n) acc = fmaf(ws[A_CRYS + g * ATOMF + n], out_W[n], acc);
    out[(long)N_NODES * ATOMF + g] = acc + out_b[0];
}

extern "C" void kernel_launch(void* const* d_in, const int* in_sizes, int n_in,
                              void* d_out, int out_size, void* d_ws, size_t ws_size,
                              hipStream_t stream) {
    const float* x         = (const float*)d_in[0];
    const float* edge_attr = (const float*)d_in[1];
    const float* lower_f   = (const float*)d_in[2];
    const int*   ei        = (const int*)  d_in[3];
    const int*   batch     = (const int*)  d_in[4];
    const float* emb_W     = (const float*)d_in[5];
    const float* emb_b     = (const float*)d_in[6];
    const float* fc_W      = (const float*)d_in[7];
    // d_in[8] = fc_b  (cancels inside BN1)
    const float* bn1_g     = (const float*)d_in[9];
    const float* bn1_b     = (const float*)d_in[10];
    const float* bn2_g     = (const float*)d_in[11];
    const float* bn2_b     = (const float*)d_in[12];
    const float* att_W     = (const float*)d_in[13];
    const float* att_b     = (const float*)d_in[14];
    const float* out_W     = (const float*)d_in[15];
    const float* out_b     = (const float*)d_in[16];

    float* ws  = (float*)d_ws;
    float* out = (float*)d_out;

    {   // K0: zero summed + accumulators
        long tot = (long)N_NODES * ATOMF + ACC_COUNT;
        int blocks = (int)((tot + 255) / 256);
        k0_init<<<blocks, 256, 0, stream>>>(ws);
    }
    k1_node_embed<<<N_NODES, 96, 0, stream>>>(x, lower_f, emb_W, emb_b, ws);
    k1b_convW<<<(WFC_HALVES + 255) / 256, 256, 0, stream>>>(fc_W, ws);
    {   // K2: WMMA GEMM pass 1 (stats)
        int blocks = (N_EDGES + EDGES_PER_BLOCK - 1) / EDGES_PER_BLOCK;
        k2_gemm_stats<<<blocks, 256, 0, stream>>>(edge_attr, ei, ws);
    }
    k3_fin1<<<1, 128, 0, stream>>>(bn1_g, bn1_b, ws);
    {   // K4: WMMA GEMM pass 2 (BN + gated msg + scatter)
        int blocks = (N_EDGES + EDGES_PER_BLOCK - 1) / EDGES_PER_BLOCK;
        k4_gemm_msg<<<blocks, 256, 0, stream>>>(edge_attr, ei, ws);
    }
    k5_stats2<<<(N_NODES + 255) / 256, 64, 0, stream>>>(ws);
    k6_fin2<<<1, 64, 0, stream>>>(bn2_g, bn2_b, ws);
    k7_node<<<N_NODES, 64, 0, stream>>>(att_W, att_b, batch, ws, out);
    k8_expsum<<<(N_NODES + 255) / 256, 256, 0, stream>>>(batch, ws);
    {
        long tot = (long)N_NODES * ATOMF;
        k9_pool<<<(int)((tot + 255) / 256), 256, 0, stream>>>(batch, ws);
    }
    k10_out<<<1, 256, 0, stream>>>(out_W, out_b, ws, out);
}
